// ARKP_Net_9646496547570
// MI455X (gfx1250) — matrix operand order repaired and checked by reading, not compiled, not run on hardware
//
#include <hip/hip_runtime.h>
#include <hip/hip_bf16.h>

// =====================  CDNA5 (gfx1250) WMMA types  =========================
typedef __attribute__((ext_vector_type(16))) _Float16     v16h;
typedef __attribute__((ext_vector_type(8)))  float        v8f;
typedef __attribute__((ext_vector_type(4)))  unsigned int u32x4;

union V16H { v16h v; u32x4 u[2]; _Float16 h[16]; };
union V8H  { u32x4 u; _Float16 h[8]; };

#define WMMA_F16(a, b, c) \
  __builtin_amdgcn_wmma_f32_16x16x32_f16(false, (a), false, (b), (short)0, (c), false, false)

static __device__ __forceinline__ v8f v8f_zero() {
  v8f z = {0.f, 0.f, 0.f, 0.f, 0.f, 0.f, 0.f, 0.f};
  return z;
}

// =====================  conv1: y = W1[64,6] @ x, + BN stats  =================
__global__ __launch_bounds__(256) void conv1_kernel(const float* __restrict__ x,
                                                    const float* __restrict__ w1,
                                                    float* __restrict__ hraw,
                                                    float* __restrict__ statsRaw) {
  __shared__ float sW[384];
  __shared__ float sSum[64], sSq[64];
  const int tid = threadIdx.x;
  for (int i = tid; i < 384; i += 256) sW[i] = w1[i];
  for (int i = tid; i < 64; i += 256) { sSum[i] = 0.f; sSq[i] = 0.f; }
  __syncthreads();
  const int idx = blockIdx.x * 256 + tid;
  const int b = idx >> 10, n = idx & 1023;
  float xv[6];
#pragma unroll
  for (int c = 0; c < 6; ++c) xv[c] = x[((size_t)b * 6 + c) * 1024 + n];
  for (int o = 0; o < 64; ++o) {
    float y = 0.f;
#pragma unroll
    for (int c = 0; c < 6; ++c) y += sW[o * 6 + c] * xv[c];
    hraw[((size_t)b * 64 + o) * 1024 + n] = y;
    float s = y, q = y * y;
#pragma unroll
    for (int off = 16; off > 0; off >>= 1) { s += __shfl_xor(s, off); q += __shfl_xor(q, off); }
    if ((tid & 31) == 0) { atomicAdd(&sSum[o], s); atomicAdd(&sSq[o], q); }
  }
  __syncthreads();
  for (int i = tid; i < 64; i += 256) {
    atomicAdd(&statsRaw[i], sSum[i]);
    atomicAdd(&statsRaw[64 + i], sSq[i]);
  }
}

// ====== finalize: raw [sum|sumsq] -> [scale|shift] at p+2C / p+3C ===========
__global__ __launch_bounds__(256) void finalize_stats(float* __restrict__ p,
                                                      const float* __restrict__ gamma,
                                                      const float* __restrict__ beta,
                                                      float invCnt, int C) {
  const int i = blockIdx.x * 256 + threadIdx.x;
  if (i < C) {
    float m   = p[i] * invCnt;
    float var = p[C + i] * invCnt - m * m;
    float inv = rsqrtf(var + 1e-5f);
    float sc  = gamma[i] * inv;
    p[2 * C + i] = sc;
    p[3 * C + i] = beta[i] - sc * m;
  }
}

// ====== apply BN+relu to conv1; emit feat0 f16 [B,N,64] point-major + xyz ===
__global__ __launch_bounds__(256) void apply_h(const float* __restrict__ hraw,
                                               const float* __restrict__ x,
                                               const float* __restrict__ bn1f,
                                               _Float16* __restrict__ feat0,
                                               float* __restrict__ xyz) {
  __shared__ float sc[64], sh[64];
  const int tid = threadIdx.x;
  for (int i = tid; i < 64; i += 256) { sc[i] = bn1f[i]; sh[i] = bn1f[64 + i]; }
  __syncthreads();
  const int idx = blockIdx.x * 256 + tid;
  const int b = idx >> 10, n = idx & 1023;
  for (int o = 0; o < 64; ++o) {
    float v = hraw[((size_t)b * 64 + o) * 1024 + n] * sc[o] + sh[o];
    v = v > 0.f ? v : 0.f;
    feat0[((size_t)b * 1024 + n) * 64 + o] = (_Float16)v;
  }
#pragma unroll
  for (int j = 0; j < 3; ++j)
    xyz[((size_t)b * 1024 + n) * 3 + j] = x[((size_t)b * 6 + j) * 1024 + n];
}

// =====================  Farthest point sampling  ============================
template <int P, int PPT>
__global__ __launch_bounds__(256) void fps_kernel(const float* __restrict__ xyz, int npoint,
                                                  int* __restrict__ fidx,
                                                  float* __restrict__ newxyz) {
  __shared__ float sx[P], sy[P], sz[P];
  __shared__ float rv[256];
  __shared__ int   ri[256];
  __shared__ float lastp[3];
  const int b = blockIdx.x, tid = threadIdx.x;
  const float* base = xyz + (size_t)b * P * 3;
  for (int p = tid; p < P; p += 256) {
    sx[p] = base[p * 3 + 0];
    sy[p] = base[p * 3 + 1];
    sz[p] = base[p * 3 + 2];
  }
  float dist[PPT];
#pragma unroll
  for (int j = 0; j < PPT; ++j) dist[j] = 1e10f;
  if (tid == 0) {
    fidx[(size_t)b * npoint] = 0;
    lastp[0] = base[0]; lastp[1] = base[1]; lastp[2] = base[2];
    newxyz[(size_t)b * npoint * 3 + 0] = base[0];
    newxyz[(size_t)b * npoint * 3 + 1] = base[1];
    newxyz[(size_t)b * npoint * 3 + 2] = base[2];
  }
  __syncthreads();
  for (int it = 1; it < npoint; ++it) {
    const float lx = lastp[0], ly = lastp[1], lz = lastp[2];
    float best = -1.f; int bi = tid;
#pragma unroll
    for (int j = 0; j < PPT; ++j) {
      int p = tid + j * 256;
      float dx = sx[p] - lx, dy = sy[p] - ly, dz = sz[p] - lz;
      float d = dx * dx + dy * dy + dz * dz;
      dist[j] = fminf(dist[j], d);
      if (dist[j] > best) { best = dist[j]; bi = p; }
    }
    rv[tid] = best; ri[tid] = bi;
    __syncthreads();
    for (int off = 128; off > 0; off >>= 1) {
      if (tid < off) {
        float ov = rv[tid + off]; int oi = ri[tid + off];
        if (ov > rv[tid] || (ov == rv[tid] && oi < ri[tid])) { rv[tid] = ov; ri[tid] = oi; }
      }
      __syncthreads();
    }
    if (tid == 0) {
      int sel = ri[0];
      fidx[(size_t)b * npoint + it] = sel;
      newxyz[((size_t)b * npoint + it) * 3 + 0] = sx[sel];
      newxyz[((size_t)b * npoint + it) * 3 + 1] = sy[sel];
      newxyz[((size_t)b * npoint + it) * 3 + 2] = sz[sel];
      lastp[0] = sx[sel]; lastp[1] = sy[sel]; lastp[2] = sz[sel];
    }
    __syncthreads();
  }
}

// =====================  kNN: 32 smallest by iterative argmin  ===============
template <int P>
__global__ __launch_bounds__(256) void knn_kernel(const float* __restrict__ pts,
                                                  const float* __restrict__ ctrs, int S,
                                                  int* __restrict__ knn) {
  __shared__ float sd[P];
  __shared__ float rv[256];
  __shared__ int   ri[256];
  const int g = blockIdx.x;
  const int b = g / S;
  const int tid = threadIdx.x;
  const float cx = ctrs[(size_t)g * 3 + 0];
  const float cy = ctrs[(size_t)g * 3 + 1];
  const float cz = ctrs[(size_t)g * 3 + 2];
  const float* base = pts + (size_t)b * P * 3;
  for (int p = tid; p < P; p += 256) {
    float dx = base[p * 3] - cx, dy = base[p * 3 + 1] - cy, dz = base[p * 3 + 2] - cz;
    sd[p] = dx * dx + dy * dy + dz * dz;
  }
  __syncthreads();
  for (int k = 0; k < 32; ++k) {
    float best = 3.0e38f; int bi = tid;
    for (int p = tid; p < P; p += 256) {
      float d = sd[p];
      if (d < best) { best = d; bi = p; }
    }
    rv[tid] = best; ri[tid] = bi;
    __syncthreads();
    for (int off = 128; off > 0; off >>= 1) {
      if (tid < off) {
        float ov = rv[tid + off]; int oi = ri[tid + off];
        if (ov < rv[tid] || (ov == rv[tid] && oi < ri[tid])) { rv[tid] = ov; ri[tid] = oi; }
      }
      __syncthreads();
    }
    if (tid == 0) { knn[(size_t)g * 32 + k] = ri[0]; sd[ri[0]] = 3.0e38f; }
    __syncthreads();
  }
}

// ===== x_skip WMMA GEMM: W2[1024,64] @ feat0[:, ::4 points] -> raw + stats ==
__global__ __launch_bounds__(256) void xskip_gemm(const _Float16* __restrict__ feat0,
                                                  const _Float16* __restrict__ W2h,
                                                  float* __restrict__ yraw,
                                                  float* __restrict__ statsRaw) {
  __shared__ float sSum[1024], sSq[1024];
  const int tid = threadIdx.x, lane = tid & 31, wave = tid >> 5;
  const int hi = lane >> 4, ln = lane & 15;
  const int b = blockIdx.x >> 3, cg = blockIdx.x & 7;
  for (int i = tid; i < 1024; i += 256) { sSum[i] = 0.f; sSq[i] = 0.f; }
  __syncthreads();
  const int col0 = cg * 32 + ln, col1 = col0 + 16;
  V16H b0[2], b1[2];
#pragma unroll
  for (int kk = 0; kk < 2; ++kk) {
    const int kc = kk * 32;
    const _Float16* p0 = feat0 + ((size_t)b * 1024 + 4 * col0) * 64 + kc + hi * 16;
    b0[kk].u[0] = *(const u32x4*)(p0);
    b0[kk].u[1] = *(const u32x4*)(p0 + 8);
    const _Float16* p1 = feat0 + ((size_t)b * 1024 + 4 * col1) * 64 + kc + hi * 16;
    b1[kk].u[0] = *(const u32x4*)(p1);
    b1[kk].u[1] = *(const u32x4*)(p1 + 8);
  }
  for (int rt = wave; rt < 64; rt += 8) {
    v8f acc0 = v8f_zero(), acc1 = v8f_zero();
#pragma unroll
    for (int kk = 0; kk < 2; ++kk) {
      const int kc = kk * 32;
      V16H a;
      const _Float16* wr = W2h + (size_t)(rt * 16 + ln) * 64 + kc;
      a.u[0] = *(const u32x4*)(wr + hi * 16);
      a.u[1] = *(const u32x4*)(wr + hi * 16 + 8);
      acc0 = WMMA_F16(a.v, b0[kk].v, acc0);
      acc1 = WMMA_F16(a.v, b1[kk].v, acc1);
    }
#pragma unroll
    for (int i = 0; i < 8; ++i) {
      const int row = rt * 16 + hi * 8 + i;
      float v0 = acc0[i], v1 = acc1[i];
      yraw[((size_t)b * 1024 + row) * 256 + col0] = v0;
      yraw[((size_t)b * 1024 + row) * 256 + col1] = v1;
      atomicAdd(&sSum[row], v0 + v1);
      atomicAdd(&sSq[row], v0 * v0 + v1 * v1);
    }
  }
  __syncthreads();
  for (int i = tid; i < 1024; i += 256) {
    atomicAdd(&statsRaw[i], sSum[i]);
    atomicAdd(&statsRaw[1024 + i], sSq[i]);
  }
}

// ===== apply BN+relu to x_skip, write into FcatT[b][n][256..1279] f16 =======
__global__ __launch_bounds__(256) void xskip_apply(const float* __restrict__ yraw,
                                                   const float* __restrict__ bn2f,
                                                   _Float16* __restrict__ FcatT) {
  const int b = blockIdx.x >> 8, n = blockIdx.x & 255;
  for (int c = threadIdx.x; c < 1024; c += 256) {
    float v = yraw[((size_t)b * 1024 + c) * 256 + n] * bn2f[c] + bn2f[1024 + c];
    v = v > 0.f ? v : 0.f;
    FcatT[((size_t)b * 256 + n) * 1280 + 256 + c] = (_Float16)v;
  }
}

// ===== grouped local_op (gather + 2 GEMMs); PASS 1/2 = stats, 3 = output ====
template <int D, int O, int C, int PASS>
__global__ __launch_bounds__(256) void local_op_pass(
    const _Float16* __restrict__ feat, int Npts, const int* __restrict__ knn,
    const int* __restrict__ fidx, int S, const _Float16* __restrict__ W1,
    const _Float16* __restrict__ W2, const float* __restrict__ bn1f,
    const float* __restrict__ bn2f, float* __restrict__ statsRaw,
    _Float16* __restrict__ outFeat, int outStride) {
  constexpr int DP = D + 8;
  constexpr int OP = O + 8;
  __shared__ _Float16 sC[C];
  __shared__ _Float16 XT[32][DP];
  __shared__ _Float16 X1T[32][OP];
  __shared__ float sSum[O];
  __shared__ float sSq[O];

  const int g = blockIdx.x;
  const int b = g / S;
  const int tid = threadIdx.x;
  const int lane = tid & 31;
  const int wave = tid >> 5;
  const int hi = lane >> 4;
  const int ln = lane & 15;

  const _Float16* featB = feat + (size_t)b * Npts * C;
  const int ctr = fidx[g];

  for (int d = tid; d < C; d += 256) sC[d] = featB[(size_t)ctr * C + d];
  if (PASS != 3)
    for (int i = tid; i < O; i += 256) { sSum[i] = 0.f; sSq[i] = 0.f; }
  __syncthreads();

  // gather X[d][k] K-major into LDS: d<C -> grouped-center, else center
  for (int e = tid; e < 32 * D; e += 256) {
    const int k = e / D;
    const int d = e - k * D;
    const int p = knn[(size_t)g * 32 + k];
    float v;
    if (d < C) v = (float)featB[(size_t)p * C + d] - (float)sC[d];
    else       v = (float)sC[d - C];
    XT[k][d] = (_Float16)v;
  }
  __syncthreads();

  // GEMM1: Z1 = W1[O,D] @ X[D,32]
  for (int ot = wave; ot < O / 16; ot += 8) {
    v8f acc0 = v8f_zero(), acc1 = v8f_zero();
#pragma unroll
    for (int kc = 0; kc < D; kc += 32) {
      V16H a, f0, f1;
      const _Float16* wr = W1 + (size_t)(ot * 16 + ln) * D + kc;
      a.u[0] = *(const u32x4*)(wr + hi * 16);
      a.u[1] = *(const u32x4*)(wr + hi * 16 + 8);
      const _Float16* x0 = &XT[ln][kc + hi * 16];
      f0.u[0] = *(const u32x4*)(x0);
      f0.u[1] = *(const u32x4*)(x0 + 8);
      const _Float16* x1 = &XT[16 + ln][kc + hi * 16];
      f1.u[0] = *(const u32x4*)(x1);
      f1.u[1] = *(const u32x4*)(x1 + 8);
      acc0 = WMMA_F16(a.v, f0.v, acc0);
      acc1 = WMMA_F16(a.v, f1.v, acc1);
    }
    if (PASS == 1) {
#pragma unroll
      for (int i = 0; i < 8; ++i) {
        const int row = ot * 16 + hi * 8 + i;
        float v0 = acc0[i], v1 = acc1[i];
        atomicAdd(&sSum[row], v0 + v1);
        atomicAdd(&sSq[row], v0 * v0 + v1 * v1);
      }
    } else {
      V8H o0, o1;
#pragma unroll
      for (int i = 0; i < 8; ++i) {
        const int row = ot * 16 + hi * 8 + i;
        const float sc = bn1f[row], sh = bn1f[O + row];
        float v0 = acc0[i] * sc + sh; v0 = v0 > 0.f ? v0 : 0.f;
        float v1 = acc1[i] * sc + sh; v1 = v1 > 0.f ? v1 : 0.f;
        o0.h[i] = (_Float16)v0;
        o1.h[i] = (_Float16)v1;
      }
      *(u32x4*)(&X1T[ln][ot * 16 + hi * 8]) = o0.u;
      *(u32x4*)(&X1T[16 + ln][ot * 16 + hi * 8]) = o1.u;
    }
  }
  __syncthreads();
  if (PASS == 1) {
    for (int i = tid; i < O; i += 256) {
      atomicAdd(&statsRaw[i], sSum[i]);
      atomicAdd(&statsRaw[O + i], sSq[i]);
    }
    return;
  }

  // GEMM2: Z2 = W2[O,O] @ X1[O,32]
  for (int ot = wave; ot < O / 16; ot += 8) {
    v8f acc0 = v8f_zero(), acc1 = v8f_zero();
#pragma unroll
    for (int kc = 0; kc < O; kc += 32) {
      V16H a, f0, f1;
      const _Float16* wr = W2 + (size_t)(ot * 16 + ln) * O + kc;
      a.u[0] = *(const u32x4*)(wr + hi * 16);
      a.u[1] = *(const u32x4*)(wr + hi * 16 + 8);
      const _Float16* x0 = &X1T[ln][kc + hi * 16];
      f0.u[0] = *(const u32x4*)(x0);
      f0.u[1] = *(const u32x4*)(x0 + 8);
      const _Float16* x1 = &X1T[16 + ln][kc + hi * 16];
      f1.u[0] = *(const u32x4*)(x1);
      f1.u[1] = *(const u32x4*)(x1 + 8);
      acc0 = WMMA_F16(a.v, f0.v, acc0);
      acc1 = WMMA_F16(a.v, f1.v, acc1);
    }
    if (PASS == 2) {
#pragma unroll
      for (int i = 0; i < 8; ++i) {
        const int row = ot * 16 + hi * 8 + i;
        float v0 = acc0[i], v1 = acc1[i];
        atomicAdd(&sSum[row], v0 + v1);
        atomicAdd(&sSq[row], v0 * v0 + v1 * v1);
      }
    } else {
      V8H om;
#pragma unroll
      for (int i = 0; i < 8; ++i) {
        const int row = ot * 16 + hi * 8 + i;
        const float sc = bn2f[row], sh = bn2f[O + row];
        float v0 = acc0[i] * sc + sh; v0 = v0 > 0.f ? v0 : 0.f;
        float v1 = acc1[i] * sc + sh; v1 = v1 > 0.f ? v1 : 0.f;
        float m = fmaxf(v0, v1);
#pragma unroll
        for (int off = 1; off < 16; off <<= 1) m = fmaxf(m, __shfl_xor(m, off));
        om.h[i] = (_Float16)m;
      }
      if (ln == 0)
        *(u32x4*)(outFeat + (size_t)g * outStride + ot * 16 + hi * 8) = om.u;
    }
  }
  if (PASS == 2) {
    __syncthreads();
    for (int i = tid; i < O; i += 256) {
      atomicAdd(&statsRaw[i], sSum[i]);
      atomicAdd(&statsRaw[O + i], sSq[i]);
    }
  }
}

// ===== final WMMA GEMM: Wf[512,1280] @ FcatT -> raw d_out + stats ===========
__global__ __launch_bounds__(256) void final_gemm(const _Float16* __restrict__ FcatT,
                                                  const _Float16* __restrict__ Wfh,
                                                  float* __restrict__ yout,
                                                  float* __restrict__ statsRaw) {
  __shared__ float sSum[512], sSq[512];
  const int tid = threadIdx.x, lane = tid & 31, wave = tid >> 5;
  const int hi = lane >> 4, ln = lane & 15;
  const int b = blockIdx.x >> 3, cg = blockIdx.x & 7;
  for (int i = tid; i < 512; i += 256) { sSum[i] = 0.f; sSq[i] = 0.f; }
  __syncthreads();
  const int col0 = cg * 32 + ln, col1 = col0 + 16;
  const _Float16* f0 = FcatT + ((size_t)b * 256 + col0) * 1280;
  const _Float16* f1 = FcatT + ((size_t)b * 256 + col1) * 1280;
  for (int rt = wave; rt < 32; rt += 8) {
    v8f acc0 = v8f_zero(), acc1 = v8f_zero();
    const _Float16* wr = Wfh + (size_t)(rt * 16 + ln) * 1280;
    for (int kc = 0; kc < 1280; kc += 32) {
      V16H a, bb0, bb1;
      a.u[0] = *(const u32x4*)(wr + kc + hi * 16);
      a.u[1] = *(const u32x4*)(wr + kc + hi * 16 + 8);
      const _Float16* q0 = f0 + kc + hi * 16;
      bb0.u[0] = *(const u32x4*)(q0);
      bb0.u[1] = *(const u32x4*)(q0 + 8);
      const _Float16* q1 = f1 + kc + hi * 16;
      bb1.u[0] = *(const u32x4*)(q1);
      bb1.u[1] = *(const u32x4*)(q1 + 8);
      acc0 = WMMA_F16(a.v, bb0.v, acc0);
      acc1 = WMMA_F16(a.v, bb1.v, acc1);
    }
#pragma unroll
    for (int i = 0; i < 8; ++i) {
      const int row = rt * 16 + hi * 8 + i;
      float v0 = acc0[i], v1 = acc1[i];
      yout[((size_t)b * 512 + row) * 256 + col0] = v0;
      yout[((size_t)b * 512 + row) * 256 + col1] = v1;
      atomicAdd(&sSum[row], v0 + v1);
      atomicAdd(&sSq[row], v0 * v0 + v1 * v1);
    }
  }
  __syncthreads();
  for (int i = tid; i < 512; i += 256) {
    atomicAdd(&statsRaw[i], sSum[i]);
    atomicAdd(&statsRaw[512 + i], sSq[i]);
  }
}

// ===== final BN + leaky relu (in place on d_out) ============================
__global__ __launch_bounds__(256) void final_apply(float* __restrict__ out,
                                                   const float* __restrict__ bnff) {
  const int i = blockIdx.x * 256 + threadIdx.x;   // 32*512*256 total
  const int o = (i >> 8) & 511;
  float v = out[i] * bnff[o] + bnff[512 + o];
  out[i] = v > 0.f ? v : 0.2f * v;
}

// ===== f32 -> f16 weight conversion =========================================
__global__ __launch_bounds__(256) void cvt_f16(const float* __restrict__ s,
                                               _Float16* __restrict__ d, int n) {
  const int i = blockIdx.x * 256 + threadIdx.x;
  if (i < n) d[i] = (_Float16)s[i];
}

// ============================================================================
extern "C" void kernel_launch(void* const* d_in, const int* in_sizes, int n_in,
                              void* d_out, int out_size, void* d_ws, size_t ws_size,
                              hipStream_t stream) {
  (void)in_sizes; (void)n_in; (void)out_size; (void)ws_size;
  const float* x    = (const float*)d_in[0];
  const float* w_c1 = (const float*)d_in[1];
  const float* g1   = (const float*)d_in[2];
  const float* b1   = (const float*)d_in[3];
  const float* w_c2 = (const float*)d_in[4];
  const float* g2   = (const float*)d_in[5];
  const float* b2   = (const float*)d_in[6];
  const float* l0w1 = (const float*)d_in[7];
  const float* l0g1 = (const float*)d_in[8];
  const float* l0b1 = (const float*)d_in[9];
  const float* l0w2 = (const float*)d_in[10];
  const float* l0g2 = (const float*)d_in[11];
  const float* l0b2 = (const float*)d_in[12];
  const float* l1w1 = (const float*)d_in[13];
  const float* l1g1 = (const float*)d_in[14];
  const float* l1b1 = (const float*)d_in[15];
  const float* l1w2 = (const float*)d_in[16];
  const float* l1g2 = (const float*)d_in[17];
  const float* l1b2 = (const float*)d_in[18];
  const float* wf   = (const float*)d_in[19];
  const float* gf   = (const float*)d_in[20];
  const float* bf   = (const float*)d_in[21];
  float* out = (float*)d_out;

  char* wsb = (char*)d_ws;
  size_t off = 0;
  auto alloc = [&](size_t bytes) -> char* {
    off = (off + 255) & ~(size_t)255;
    char* p = wsb + off;
    off += bytes;
    return p;
  };

  // stats: each BN set = 4*C floats [sum|sumsq|scale|shift]
  float* stats  = (float*)alloc(9472 * sizeof(float));
  float* st_bn1 = stats;                 // C=64
  float* st_bn2 = st_bn1 + 4 * 64;       // C=1024
  float* st_l0a = st_bn2 + 4 * 1024;     // C=128
  float* st_l0b = st_l0a + 4 * 128;      // C=128
  float* st_l1a = st_l0b + 4 * 128;      // C=256
  float* st_l1b = st_l1a + 4 * 256;      // C=256
  float* st_f   = st_l1b + 4 * 256;      // C=512

  _Float16* w2h   = (_Float16*)alloc((size_t)1024 * 64 * 2);
  _Float16* l0w1h = (_Float16*)alloc((size_t)128 * 128 * 2);
  _Float16* l0w2h = (_Float16*)alloc((size_t)128 * 128 * 2);
  _Float16* l1w1h = (_Float16*)alloc((size_t)256 * 256 * 2);
  _Float16* l1w2h = (_Float16*)alloc((size_t)256 * 256 * 2);
  _Float16* wfh   = (_Float16*)alloc((size_t)512 * 1280 * 2);
  float*    hraw  = (float*)alloc((size_t)32 * 64 * 1024 * 4);
  _Float16* feat0 = (_Float16*)alloc((size_t)32 * 1024 * 64 * 2);
  float*    xyz   = (float*)alloc((size_t)32 * 1024 * 3 * 4);
  int*      fidx0 = (int*)alloc((size_t)32 * 512 * 4);
  float*    nx0   = (float*)alloc((size_t)32 * 512 * 3 * 4);
  int*      knn0  = (int*)alloc((size_t)32 * 512 * 32 * 4);
  int*      fidx1 = (int*)alloc((size_t)32 * 256 * 4);
  float*    nx1   = (float*)alloc((size_t)32 * 256 * 3 * 4);
  int*      knn1  = (int*)alloc((size_t)32 * 256 * 32 * 4);
  _Float16* feat1 = (_Float16*)alloc((size_t)32 * 512 * 128 * 2);
  float*    xskraw = (float*)alloc((size_t)32 * 1024 * 256 * 4);
  _Float16* FcatT  = (_Float16*)alloc((size_t)32 * 256 * 1280 * 2);

  hipMemsetAsync(stats, 0, 9472 * sizeof(float), stream);

  cvt_f16<<<256, 256, 0, stream>>>(w_c2, w2h, 65536);
  cvt_f16<<<64, 256, 0, stream>>>(l0w1, l0w1h, 16384);
  cvt_f16<<<64, 256, 0, stream>>>(l0w2, l0w2h, 16384);
  cvt_f16<<<256, 256, 0, stream>>>(l1w1, l1w1h, 65536);
  cvt_f16<<<256, 256, 0, stream>>>(l1w2, l1w2h, 65536);
  cvt_f16<<<2560, 256, 0, stream>>>(wf, wfh, 655360);

  conv1_kernel<<<128, 256, 0, stream>>>(x, w_c1, hraw, st_bn1);
  finalize_stats<<<1, 256, 0, stream>>>(st_bn1, g1, b1, 1.0f / 32768.f, 64);
  apply_h<<<128, 256, 0, stream>>>(hraw, x, st_bn1 + 128, feat0, xyz);

  fps_kernel<1024, 4><<<32, 256, 0, stream>>>(xyz, 512, fidx0, nx0);
  fps_kernel<512, 2><<<32, 256, 0, stream>>>(nx0, 256, fidx1, nx1);
  knn_kernel<1024><<<32 * 512, 256, 0, stream>>>(xyz, nx0, 512, knn0);
  knn_kernel<512><<<32 * 256, 256, 0, stream>>>(nx0, nx1, 256, knn1);

  xskip_gemm<<<256, 256, 0, stream>>>(feat0, w2h, xskraw, st_bn2);
  finalize_stats<<<4, 256, 0, stream>>>(st_bn2, g2, b2, 1.0f / 8192.f, 1024);
  xskip_apply<<<32 * 256, 256, 0, stream>>>(xskraw, st_bn2 + 2048, FcatT);

  // stage 0: 16384 groups, D=128, O=128, C=64
  local_op_pass<128, 128, 64, 1><<<16384, 256, 0, stream>>>(
      feat0, 1024, knn0, fidx0, 512, l0w1h, l0w2h, nullptr, nullptr, st_l0a, nullptr, 0);
  finalize_stats<<<1, 256, 0, stream>>>(st_l0a, l0g1, l0b1, 1.0f / 524288.f, 128);
  local_op_pass<128, 128, 64, 2><<<16384, 256, 0, stream>>>(
      feat0, 1024, knn0, fidx0, 512, l0w1h, l0w2h, st_l0a + 256, nullptr, st_l0b, nullptr, 0);
  finalize_stats<<<1, 256, 0, stream>>>(st_l0b, l0g2, l0b2, 1.0f / 524288.f, 128);
  local_op_pass<128, 128, 64, 3><<<16384, 256, 0, stream>>>(
      feat0, 1024, knn0, fidx0, 512, l0w1h, l0w2h, st_l0a + 256, st_l0b + 256, nullptr, feat1, 128);

  // stage 1: 8192 groups, D=256, O=256, C=128
  local_op_pass<256, 256, 128, 1><<<8192, 256, 0, stream>>>(
      feat1, 512, knn1, fidx1, 256, l1w1h, l1w2h, nullptr, nullptr, st_l1a, nullptr, 0);
  finalize_stats<<<1, 256, 0, stream>>>(st_l1a, l1g1, l1b1, 1.0f / 262144.f, 256);
  local_op_pass<256, 256, 128, 2><<<8192, 256, 0, stream>>>(
      feat1, 512, knn1, fidx1, 256, l1w1h, l1w2h, st_l1a + 512, nullptr, st_l1b, nullptr, 0);
  finalize_stats<<<1, 256, 0, stream>>>(st_l1b, l1g2, l1b2, 1.0f / 262144.f, 256);
  local_op_pass<256, 256, 128, 3><<<8192, 256, 0, stream>>>(
      feat1, 512, knn1, fidx1, 256, l1w1h, l1w2h, st_l1a + 512, st_l1b + 512, nullptr, FcatT, 1280);

  final_gemm<<<256, 256, 0, stream>>>(FcatT, wfh, out, st_f);
  finalize_stats<<<2, 256, 0, stream>>>(st_f, gf, bf, 1.0f / 8192.f, 512);
  final_apply<<<16384, 256, 0, stream>>>(out, st_f + 1024);
}
